// ModelSIM_25185688224516
// MI455X (gfx1250) — matrix operand order repaired and checked by reading, not compiled, not run on hardware
//
#include <hip/hip_runtime.h>

typedef __attribute__((ext_vector_type(16))) __bf16 v16bf;
typedef __attribute__((ext_vector_type(8)))  __bf16 v8bf;
typedef __attribute__((ext_vector_type(8)))  float  v8f;
typedef __attribute__((ext_vector_type(4)))  unsigned int u32x4;
typedef __attribute__((ext_vector_type(8)))  int i32x8;
typedef __attribute__((ext_vector_type(4)))  int i32x4;

#define KDIM 128

// ---------------------------------------------------------------------------
// Utility: grid-stride zero fill (graph-capture safe, no hipMemset)
// ---------------------------------------------------------------------------
__global__ void zero_kernel(float* __restrict__ p, size_t n) {
    size_t i = (size_t)blockIdx.x * blockDim.x + threadIdx.x;
    size_t stride = (size_t)gridDim.x * blockDim.x;
    for (; i < n; i += stride) p[i] = 0.0f;
}

// ---------------------------------------------------------------------------
// f32 -> bf16 conversion, 8 elems / iter (16B vector stores). n8 = n/8.
// ---------------------------------------------------------------------------
__global__ void cvt_bf16_kernel(const float* __restrict__ in,
                                __bf16* __restrict__ out, size_t n8) {
    size_t i = (size_t)blockIdx.x * blockDim.x + threadIdx.x;
    size_t stride = (size_t)gridDim.x * blockDim.x;
    for (; i < n8; i += stride) {
        const float* p = in + i * 8;
        v8bf o;
        #pragma unroll
        for (int j = 0; j < 8; ++j) o[j] = (__bf16)p[j];
        *(v8bf*)(out + i * 8) = o;
    }
}

// ---------------------------------------------------------------------------
// Degree count (float atomics into L2-resident 400KB array)
// ---------------------------------------------------------------------------
__global__ void degree_kernel(const long long* __restrict__ dst, int E,
                              float* __restrict__ deg) {
    int e = blockIdx.x * blockDim.x + threadIdx.x;
    if (e < E) atomicAdd(&deg[(int)dst[e]], 1.0f);
}

__global__ void rsqrt_kernel(const float* __restrict__ deg,
                             float* __restrict__ dis, int N) {
    int i = blockIdx.x * blockDim.x + threadIdx.x;
    if (i < N) dis[i] = rsqrtf(deg[i] + 1.0f);
}

// ---------------------------------------------------------------------------
// Tiled GEMM: H[M,N] = A[M,128](bf16) @ W[128,N](f32->bf16), WMMA f32 acc.
// Block = 256 threads (8 waves); A tile (128x128 bf16) staged into LDS by the
// Tensor Data Mover (tensor_load_to_lds) with descriptor-driven row padding
// (64 DWORDs data + 4 DWORDs pad -> 136-elem LDS row stride) and OOB zero-fill
// for the ragged last tile (tensor_dim1 = M - m0). W transposed by ALUs in
// parallel with the DMA.
// ---------------------------------------------------------------------------
template<int NCOLS>
__global__ __launch_bounds__(256)
void gemm_bf16_wmma(const __bf16* __restrict__ A, const float* __restrict__ W,
                    float* __restrict__ H, int M) {
    constexpr int AS_STRIDE = 136;  // bf16 elems; 272B row -> 16B aligned slots
    constexpr int WT_STRIDE = 144;  // bf16 elems; 288B row -> 32B aligned slots
    extern __shared__ __align__(1024) char smem[];
    __bf16* As = (__bf16*)smem;                                  // 128*136*2 B
    __bf16* Wt = (__bf16*)(smem + 128 * AS_STRIDE * 2);          // NCOLS*144*2 B

    const int tid = threadIdx.x;
    const int m0  = blockIdx.x * 128;

    // ---- Wave 0: issue TDM load of A tile (global bf16 -> LDS, padded) ----
    if (tid < 32) {
        unsigned long long gaddr = (unsigned long long)(const void*)(A + (size_t)m0 * KDIM);
        unsigned int lds_base = (unsigned int)(size_t)(void*)smem; // low 32b = LDS offset
        int rows = M - m0;                     // OOB rows read as zero
        u32x4 g0;
        g0[0] = 1u;                                            // count=1, user D#
        g0[1] = lds_base;                                      // lds_addr
        g0[2] = (unsigned int)(gaddr & 0xFFFFFFFFu);           // global_addr[31:0]
        g0[3] = (unsigned int)((gaddr >> 32) & 0x1FFFFFFu)     // global_addr[56:32]
              | (2u << 30);                                    // type = 2 (image)
        i32x8 g1;
        g1[0] = (1 << 16)        // data_size = 1 -> 2 bytes
              | (1 << 20)        // pad_enable
              | (5 << 22)        // pad_interval: 64 DWORDs (=128 bf16 row)
              | (3 << 25);       // pad_amount: 4 DWORDs (=16B -> stride 136)
        g1[1] = (KDIM & 0xFFFF) << 16;                  // tensor_dim0 lo
        g1[2] = (rows & 0xFFFF) << 16;                  // dim0 hi=0 | tensor_dim1 lo
        g1[3] = ((rows >> 16) & 0xFFFF)                 // tensor_dim1 hi
              | (KDIM << 16);                           // tile_dim0 = 128
        g1[4] = 128;                                    // tile_dim1 = 128 rows
        g1[5] = KDIM;                                   // tensor_dim0_stride lo
        g1[6] = 0;
        g1[7] = 0;
        i32x4 gz  = {0, 0, 0, 0};                       // groups 2/3: 2-D tensor
        i32x8 gz8 = {0, 0, 0, 0, 0, 0, 0, 0};          // clang-23 extra group
        __builtin_amdgcn_tensor_load_to_lds(g0, g1, gz, gz, gz8, 0);
    }

    // ---- All threads: stage W transposed (f32 -> bf16), overlaps the DMA ----
    for (int idx = tid; idx < KDIM * NCOLS; idx += 256) {
        int k = idx / NCOLS;
        int n = idx % NCOLS;
        Wt[n * WT_STRIDE + k] = (__bf16)W[idx];
    }

    if (tid < 32) __builtin_amdgcn_s_wait_tensorcnt(0);
    __syncthreads();

    const int wave  = tid >> 5;
    const int lane  = tid & 31;
    const int lhalf = lane >> 4;   // 0: lanes 0-15, 1: lanes 16-31
    const int l16   = lane & 15;
    const int rbase = wave * 16;
    const int arow  = rbase + l16;
    const int akb   = lhalf * 16;

    v8f acc[NCOLS / 16];
    #pragma unroll
    for (int t = 0; t < NCOLS / 16; ++t)
        acc[t] = (v8f){0.f, 0.f, 0.f, 0.f, 0.f, 0.f, 0.f, 0.f};

    #pragma unroll
    for (int kt = 0; kt < KDIM / 32; ++kt) {
        const int k0 = kt * 32;
        // A lane layout: K = akb + {0..7} and akb + {8..15}
        v8bf alo = *(const v8bf*)&As[arow * AS_STRIDE + k0 + akb];
        v8bf ahi = *(const v8bf*)&As[arow * AS_STRIDE + k0 + akb + 8];
        v16bf a = __builtin_shufflevector(alo, ahi,
            0, 1, 2, 3, 4, 5, 6, 7, 8, 9, 10, 11, 12, 13, 14, 15);
        #pragma unroll
        for (int nt = 0; nt < NCOLS / 16; ++nt) {
            const int ncol = nt * 16 + l16;
            const int bk   = k0 + lhalf * 16;  // lanes 0-15: K 0..15; 16-31: 16..31
            v16bf b = *(const v16bf*)&Wt[ncol * WT_STRIDE + bk];
            acc[nt] = __builtin_amdgcn_wmma_f32_16x16x32_bf16(
                false, a, false, b, (short)0, acc[nt], false, false);
        }
    }

    // D layout: VGPR r holds row (rbase + r + lhalf*8), col (nt*16 + l16)
    #pragma unroll
    for (int nt = 0; nt < NCOLS / 16; ++nt) {
        #pragma unroll
        for (int r = 0; r < 8; ++r) {
            int row = m0 + rbase + r + lhalf * 8;
            int col = nt * 16 + l16;
            if (row < M) H[(size_t)row * NCOLS + col] = acc[nt][r];
        }
    }
}

// ---------------------------------------------------------------------------
// Edge scatter: wave per edge; lanes do F/32 coalesced f32 atomics each.
// H rows and AGG are L2-resident (51MB each << 192MB L2).
// ---------------------------------------------------------------------------
template<int F>
__global__ __launch_bounds__(256)
void scatter_kernel(const float* __restrict__ h,
                    const long long* __restrict__ src,
                    const long long* __restrict__ dst,
                    const float* __restrict__ dis,
                    float* __restrict__ agg, int E) {
    int wid  = (int)(((size_t)blockIdx.x * blockDim.x + threadIdx.x) >> 5);
    int lane = threadIdx.x & 31;
    if (wid >= E) return;
    int s = (int)src[wid];
    int d = (int)dst[wid];
    float c = dis[s] * dis[d];
    const float* hp = h + (size_t)s * F;
    float* ap = agg + (size_t)d * F;
    #pragma unroll
    for (int j = 0; j < F / 32; ++j) {
        int f = lane + j * 32;
        atomicAdd(&ap[f], hp[f] * c);
    }
}

// ---------------------------------------------------------------------------
// Finalize (in place on agg): out = relu(agg + h * dis^2 + bias)
// ---------------------------------------------------------------------------
template<int F>
__global__ __launch_bounds__(256)
void finalize_kernel(const float* __restrict__ h,
                     const float* __restrict__ dis,
                     const float* __restrict__ bias,
                     float* __restrict__ agg, int N) {
    size_t i = (size_t)blockIdx.x * blockDim.x + threadIdx.x;
    if (i >= (size_t)N * F) return;
    int node = (int)(i / F);
    int f    = (int)(i % F);
    float dd = dis[node];
    float v  = agg[i] + h[i] * dd * dd + bias[f];
    agg[i] = v > 0.f ? v : 0.f;
}

// ---------------------------------------------------------------------------
extern "C" void kernel_launch(void* const* d_in, const int* in_sizes, int n_in,
                              void* d_out, int out_size, void* d_ws, size_t ws_size,
                              hipStream_t stream) {
    const float*     x  = (const float*)d_in[0];
    const long long* ei = (const long long*)d_in[1];   // int64 [2, E]
    const float*     W1 = (const float*)d_in[2];
    const float*     b1 = (const float*)d_in[3];
    const float*     W2 = (const float*)d_in[4];
    const float*     b2 = (const float*)d_in[5];

    const int N = in_sizes[0] / 128;     // 100000
    const int E = in_sizes[1] / 2;       // 1600000
    const long long* src = ei;
    const long long* dst = ei + E;

    // Workspace layout
    char*  ws  = (char*)d_ws;
    float* deg = (float*)ws;                                    // N floats
    float* dis = deg + N;                                       // N floats
    size_t hoff = ((size_t)2 * N * sizeof(float) + 1023) & ~(size_t)1023;
    float*  H1   = (float*)(ws + hoff);                         // N*128 f32
    float*  AGG1 = H1 + (size_t)N * 128;                        // N*128 f32 (-> out1)
    __bf16* Abf  = (__bf16*)(AGG1 + (size_t)N * 128);           // N*128 bf16 (shared)
    float*  H2   = H1;                                          // reuse H1 space
    float*  OUT2 = (float*)d_out;                               // agg2 + final in place

    const int T = 256;
    const int zb = 4096;
    const size_t lds1 = 128 * 136 * 2 + 128 * 144 * 2;          // gemm<128> LDS bytes
    const size_t lds2 = 128 * 136 * 2 + 64 * 144 * 2;           // gemm<64>  LDS bytes

    // degree -> deg_inv_sqrt
    zero_kernel<<<zb, T, 0, stream>>>(deg, (size_t)N);
    degree_kernel<<<(E + T - 1) / T, T, 0, stream>>>(dst, E, deg);
    rsqrt_kernel<<<(N + T - 1) / T, T, 0, stream>>>(deg, dis, N);

    // ---- Layer 1 (128 -> 128) ----
    cvt_bf16_kernel<<<zb, T, 0, stream>>>(x, Abf, (size_t)N * 128 / 8);
    gemm_bf16_wmma<128><<<(N + 127) / 128, T, lds1, stream>>>(Abf, W1, H1, N);
    zero_kernel<<<zb, T, 0, stream>>>(AGG1, (size_t)N * 128);
    scatter_kernel<128><<<(E + 7) / 8, T, 0, stream>>>(H1, src, dst, dis, AGG1, E);
    finalize_kernel<128><<<((size_t)N * 128 + T - 1) / T, T, 0, stream>>>(
        H1, dis, b1, AGG1, N);

    // ---- Layer 2 (128 -> 64) ----
    cvt_bf16_kernel<<<zb, T, 0, stream>>>(AGG1, Abf, (size_t)N * 128 / 8);
    gemm_bf16_wmma<64><<<(N + 127) / 128, T, lds2, stream>>>(Abf, W2, H2, N);
    zero_kernel<<<zb, T, 0, stream>>>(OUT2, (size_t)N * 64);
    scatter_kernel<64><<<(E + 7) / 8, T, 0, stream>>>(H2, src, dst, dis, OUT2, E);
    finalize_kernel<64><<<((size_t)N * 64 + T - 1) / T, T, 0, stream>>>(
        H2, dis, b2, OUT2, N);
}
